// Encoder_39170101739585
// MI455X (gfx1250) — matrix operand order repaired and checked by reading, not compiled, not run on hardware
//
#include <hip/hip_runtime.h>
#include <hip/hip_bf16.h>

// ---------------------------------------------------------------------------
// LFADS bidirectional GRU encoder for gfx1250 (MI455X).
// bf16 WMMA (f32 accum) for all matmuls; recurrence parallelized over batch.
// ---------------------------------------------------------------------------

typedef __bf16 bf16_t;
typedef __attribute__((ext_vector_type(16))) __bf16 v16bf;
typedef __attribute__((ext_vector_type(8)))  float  v8f;

#define WMMA_BF16(a, b, c) \
  __builtin_amdgcn_wmma_f32_16x16x32_bf16(false, (a), false, (b), (short)0, (c), false, false)

constexpr int Bb = 512;   // batch
constexpr int Tt = 512;   // time
constexpr int Dd = 256;   // input dim
constexpr int Hh = 128;   // GRU hidden (IC_ENC == CI_ENC)
constexpr int Gg = 384;   // 3*H gate rows
constexpr float CLIPV = 5.0f;

// workspace layout (bytes)
constexpr size_t OFF_ICWIH = 0;                              // 2*G*D bf16
constexpr size_t OFF_CIWIH = OFF_ICWIH + (size_t)2*Gg*Dd*2;  // 393216
constexpr size_t OFF_ICWHH = OFF_CIWIH + (size_t)2*Gg*Dd*2;  // 786432
constexpr size_t OFF_CIWHH = OFF_ICWHH + (size_t)2*Gg*Hh*2;  // 983040
constexpr size_t OFF_HFIN  = OFF_CIWHH + (size_t)2*Gg*Hh*2;  // 1179648 (float[2*B*H])

// ---- A-matrix fragment (16x32 bf16, MxK) from an fp32 source -------------
// Layout: lane<16 holds M=lane, K=k0+0..15;
//         lane>=16 holds M=lane-16, K=k0+16..31.
__device__ inline v16bf load_a_f32(const float* base, int ld, int k0, int lane) {
  const int M   = lane & 15;
  const int off = (lane < 16) ? 0 : 16;
  const float* p = base + (size_t)M * ld + (k0 + off);
  const float4 u0 = ((const float4*)p)[0];
  const float4 u1 = ((const float4*)p)[1];
  const float4 u2 = ((const float4*)(p + 8))[0];
  const float4 u3 = ((const float4*)(p + 8))[1];
  v16bf a;
  a[0]  = (bf16_t)u0.x; a[1]  = (bf16_t)u0.y; a[2]  = (bf16_t)u0.z; a[3]  = (bf16_t)u0.w;
  a[4]  = (bf16_t)u1.x; a[5]  = (bf16_t)u1.y; a[6]  = (bf16_t)u1.z; a[7]  = (bf16_t)u1.w;
  a[8]  = (bf16_t)u2.x; a[9]  = (bf16_t)u2.y; a[10] = (bf16_t)u2.z; a[11] = (bf16_t)u2.w;
  a[12] = (bf16_t)u3.x; a[13] = (bf16_t)u3.y; a[14] = (bf16_t)u3.z; a[15] = (bf16_t)u3.w;
  return a;
}

// ---- B-matrix fragment (32x16 bf16, KxN) from row-major W[g][k] ----------
// B[k][n] = W[n0+n][k0+k]; lane<16 -> N=lane, K=k0..k0+15; lane>=16 -> K=k0+16..31.
// 16 contiguous bf16 = one aligned 32-byte vector load.
__device__ inline v16bf load_b_bf16(const bf16_t* W, int ld, int n0, int k0, int lane) {
  const int N    = lane & 15;
  const int koff = (lane < 16) ? 0 : 16;
  return *(const v16bf*)(W + (size_t)(n0 + N) * ld + (k0 + koff));
}

// ---- D-matrix (16x16 f32) store to LDS -----------------------------------
__device__ inline void store_d(float* buf, int ldb, int n0, v8f c, int lane) {
  const int N  = lane & 15;
  const int Mb = (lane < 16) ? 0 : 8;
#pragma unroll
  for (int r = 0; r < 8; ++r) buf[(size_t)(Mb + r) * ldb + (n0 + N)] = c[r];
}

// ---------------------------------------------------------------------------
// Main fused bi-GRU kernel. grid.x = 2(enc) * 2(dir) * 32(batch tiles) = 128.
// ---------------------------------------------------------------------------
__global__ __launch_bounds__(256) void lfads_bigru_kernel(
    const float* __restrict__ data,
    const float* __restrict__ ic_h0, const float* __restrict__ ci_h0,
    const bf16_t* __restrict__ icWih, const bf16_t* __restrict__ ciWih,
    const bf16_t* __restrict__ icWhh, const bf16_t* __restrict__ ciWhh,
    const float* __restrict__ ic_bih, const float* __restrict__ ic_bhh,
    const float* __restrict__ ci_bih, const float* __restrict__ ci_bhh,
    float* __restrict__ ciOut,   // d_out + 2*B*64  ([2B, T, H])
    float* __restrict__ hfin)    // ws float[2][B][H]
{
  __shared__ float sPx [16][388];  // x projection, g = 0..383
  __shared__ float sPzr[16][260];  // h@Wzr^T; later [z | r*h]
  __shared__ float sPn [16][132];  // (r*h)@Wn^T
  __shared__ float sH  [16][132];  // hidden state (fp32)

  const int tid  = threadIdx.x;
  const int lane = tid & 31;
  const int wid  = tid >> 5;
  const int unit = blockIdx.x;
  const int tb   = unit & 31;
  const int dir  = (unit >> 5) & 1;
  const int enc  = unit >> 6;        // 0 = IC encoder, 1 = CI encoder
  const int b0   = tb * 16;

  const bf16_t* Wih = (enc ? ciWih : icWih) + (size_t)dir * Gg * Dd;
  const bf16_t* Whh = (enc ? ciWhh : icWhh) + (size_t)dir * Gg * Hh;
  const float*  bih = (enc ? ci_bih : ic_bih) + (size_t)dir * Gg;
  const float*  bhh = (enc ? ci_bhh : ic_bhh) + (size_t)dir * Gg;
  const float*  h0  = (enc ? ci_h0  : ic_h0 ) + (size_t)dir * Hh;

  for (int idx = tid; idx < 16 * Hh; idx += blockDim.x)
    sH[idx >> 7][idx & 127] = h0[idx & 127];
  __syncthreads();

  for (int s = 0; s < Tt; ++s) {
    const int t = dir ? (Tt - 1 - s) : s;
    const float* xbase = data + (size_t)b0 * Tt * Dd + (size_t)t * Dd;

    // ---- GEMM 1: Px[16,384] = X_t[16,256] @ Wih^T (24 N-tiles, 8 K-steps)
    {
      v8f acc[3] = {};
#pragma unroll
      for (int kt = 0; kt < 8; ++kt) {
        v16bf a = load_a_f32(xbase, Tt * Dd, 32 * kt, lane);
#pragma unroll
        for (int i = 0; i < 3; ++i) {
          v16bf b = load_b_bf16(Wih, Dd, 16 * (wid + 8 * i), 32 * kt, lane);
          acc[i] = WMMA_BF16(a, b, acc[i]);
        }
      }
#pragma unroll
      for (int i = 0; i < 3; ++i) store_d(&sPx[0][0], 388, 16 * (wid + 8 * i), acc[i], lane);
    }
    // ---- GEMM 2: Pzr[16,256] = H[16,128] @ Wzr^T (16 N-tiles, 4 K-steps)
    {
      v8f acc[2] = {};
#pragma unroll
      for (int kt = 0; kt < 4; ++kt) {
        v16bf a = load_a_f32(&sH[0][0], 132, 32 * kt, lane);
#pragma unroll
        for (int i = 0; i < 2; ++i) {
          v16bf b = load_b_bf16(Whh, Hh, 16 * (wid + 8 * i), 32 * kt, lane);
          acc[i] = WMMA_BF16(a, b, acc[i]);
        }
      }
#pragma unroll
      for (int i = 0; i < 2; ++i) store_d(&sPzr[0][0], 260, 16 * (wid + 8 * i), acc[i], lane);
    }
    __syncthreads();

    // ---- EW1: z = sig(xz+hz), r = sig(xr+hr); stash z and r*h in sPzr
    for (int idx = tid; idx < 16 * Hh; idx += blockDim.x) {
      const int bl = idx >> 7, j = idx & 127;
      const float xz = sPx[bl][j]       + bih[j];
      const float xr = sPx[bl][Hh + j]  + bih[Hh + j];
      const float hz = sPzr[bl][j]      + bhh[j];
      const float hr = sPzr[bl][Hh + j] + bhh[Hh + j];
      const float h  = sH[bl][j];
      const float z  = 1.f / (1.f + __expf(-(xz + hz)));
      const float r  = 1.f / (1.f + __expf(-(xr + hr)));
      sPzr[bl][j]      = z;      // keep for EW2
      sPzr[bl][Hh + j] = r * h;  // A-matrix for GEMM 3 (reference: (r*h) @ Wn^T)
    }
    __syncthreads();

    // ---- GEMM 3: Pn[16,128] = (r*h) @ Wn^T (8 N-tiles, 4 K-steps)
    {
      v8f acc = {};
#pragma unroll
      for (int kt = 0; kt < 4; ++kt) {
        v16bf a = load_a_f32(&sPzr[0][Hh], 260, 32 * kt, lane);
        v16bf b = load_b_bf16(Whh + (size_t)2 * Hh * Hh, Hh, 16 * wid, 32 * kt, lane);
        acc = WMMA_BF16(a, b, acc);
      }
      store_d(&sPn[0][0], 132, 16 * wid, acc, lane);
    }
    __syncthreads();

    // ---- EW2: n = tanh(xn + Pn + bhn); h = clip(z*h + (1-z)*n); emit CI
    for (int idx = tid; idx < 16 * Hh; idx += blockDim.x) {
      const int bl = idx >> 7, j = idx & 127;
      const float xn = sPx[bl][2 * Hh + j] + bih[2 * Hh + j];
      const float hn = sPn[bl][j]          + bhh[2 * Hh + j];
      const float z  = sPzr[bl][j];
      const float h  = sH[bl][j];
      const float n  = tanhf(xn + hn);
      float hN = z * h + (1.f - z) * n;
      hN = fminf(fmaxf(hN, -CLIPV), CLIPV);
      sH[bl][j] = hN;
      if (enc == 1) {
        const int b = b0 + bl;
        if (dir == 0) {                       // fwd, CI_LAG=1: out[t] = h(t-1)
          const int to = s + 1;
          if (to < Tt) ciOut[((size_t)b * Tt + to) * Hh + j] = hN;
        } else {                              // bwd: out[T-2-s] = h(step s)
          const int to = Tt - 2 - s;
          if (to >= 0) ciOut[((size_t)(Bb + b) * Tt + to) * Hh + j] = hN;
        }
      }
    }
    __syncthreads();
  }

  if (enc == 0) {  // IC encoder: only final hidden states matter
    for (int idx = tid; idx < 16 * Hh; idx += blockDim.x) {
      const int bl = idx >> 7, j = idx & 127;
      hfin[((size_t)dir * Bb + (b0 + bl)) * Hh + j] = sH[bl][j];
    }
  }
}

// --------------------------- helper kernels --------------------------------
__global__ void cvt_f32_bf16_kernel(const float* __restrict__ s, bf16_t* __restrict__ d, int n) {
  int i = blockIdx.x * blockDim.x + threadIdx.x;
  if (i < n) d[i] = (bf16_t)s[i];
}

__global__ void zero_ci_pads_kernel(float* __restrict__ ciOut) {
  int i = blockIdx.x * blockDim.x + threadIdx.x;   // 2*B*H lanes
  if (i >= 2 * Bb * Hh) return;
  const int half = i / (Bb * Hh);
  const int b    = (i / Hh) % Bb;
  const int j    = i % Hh;
  if (half == 0) ciOut[((size_t)b * Tt + 0) * Hh + j] = 0.f;                   // fwd t=0
  else           ciOut[((size_t)(Bb + b) * Tt + (Tt - 1)) * Hh + j] = 0.f;     // bwd t=T-1
}

__global__ void ic_linear_kernel(const float* __restrict__ hfin,
                                 const float* __restrict__ Wl,
                                 const float* __restrict__ bl,
                                 float* __restrict__ out) {
  int i = blockIdx.x * blockDim.x + threadIdx.x;   // B * 128
  if (i >= Bb * 128) return;
  const int b = i >> 7, jp = i & 127;
  const float* hf = hfin + (size_t)b * Hh;                 // forward final h
  const float* hb = hfin + (size_t)(Bb + b) * Hh;          // backward final h
  const float* w  = Wl + (size_t)jp * (2 * Hh);
  float acc = bl[jp];
#pragma unroll 4
  for (int k = 0; k < Hh; ++k) acc += w[k] * hf[k];
#pragma unroll 4
  for (int k = 0; k < Hh; ++k) acc += w[Hh + k] * hb[k];
  if (jp < 64) out[(size_t)b * 64 + jp] = acc;                       // ic_mean
  else         out[(size_t)Bb * 64 + (size_t)b * 64 + (jp - 64)] = expf(acc);  // ic_std
}

// --------------------------------- launch ----------------------------------
extern "C" void kernel_launch(void* const* d_in, const int* in_sizes, int n_in,
                              void* d_out, int out_size, void* d_ws, size_t ws_size,
                              hipStream_t stream) {
  (void)in_sizes; (void)n_in; (void)out_size; (void)ws_size;
  const float* data   = (const float*)d_in[0];
  const float* ic_h0  = (const float*)d_in[1];
  const float* ci_h0  = (const float*)d_in[2];
  const float* ic_Wih = (const float*)d_in[3];
  const float* ic_bih = (const float*)d_in[4];
  const float* ic_Whh = (const float*)d_in[5];
  const float* ic_bhh = (const float*)d_in[6];
  const float* ci_Wih = (const float*)d_in[7];
  const float* ci_bih = (const float*)d_in[8];
  const float* ci_Whh = (const float*)d_in[9];
  const float* ci_bhh = (const float*)d_in[10];
  const float* Wl     = (const float*)d_in[11];
  const float* bl     = (const float*)d_in[12];

  char* wsb = (char*)d_ws;
  bf16_t* icWih = (bf16_t*)(wsb + OFF_ICWIH);
  bf16_t* ciWih = (bf16_t*)(wsb + OFF_CIWIH);
  bf16_t* icWhh = (bf16_t*)(wsb + OFF_ICWHH);
  bf16_t* ciWhh = (bf16_t*)(wsb + OFF_CIWHH);
  float*  hfin  = (float*)(wsb + OFF_HFIN);

  float* out   = (float*)d_out;
  float* ciOut = out + (size_t)2 * Bb * 64;   // after ic_mean + ic_std

  const int nWih = 2 * Gg * Dd, nWhh = 2 * Gg * Hh;
  cvt_f32_bf16_kernel<<<(nWih + 255) / 256, 256, 0, stream>>>(ic_Wih, icWih, nWih);
  cvt_f32_bf16_kernel<<<(nWih + 255) / 256, 256, 0, stream>>>(ci_Wih, ciWih, nWih);
  cvt_f32_bf16_kernel<<<(nWhh + 255) / 256, 256, 0, stream>>>(ic_Whh, icWhh, nWhh);
  cvt_f32_bf16_kernel<<<(nWhh + 255) / 256, 256, 0, stream>>>(ci_Whh, ciWhh, nWhh);

  zero_ci_pads_kernel<<<(2 * Bb * Hh + 255) / 256, 256, 0, stream>>>(ciOut);

  lfads_bigru_kernel<<<128, 256, 0, stream>>>(
      data, ic_h0, ci_h0, icWih, ciWih, icWhh, ciWhh,
      ic_bih, ic_bhh, ci_bih, ci_bhh, ciOut, hfin);

  ic_linear_kernel<<<(Bb * 128 + 255) / 256, 256, 0, stream>>>(hfin, Wl, bl, out);
}
